// DenseMissing_11802570129982
// MI455X (gfx1250) — matrix operand order repaired and checked by reading, not compiled, not run on hardware
//
#include <hip/hip_runtime.h>
#include <hip/hip_bf16.h>
#include <math.h>

typedef __attribute__((ext_vector_type(16))) __bf16       v16bf;
typedef __attribute__((ext_vector_type(8)))  float        v8f;
typedef __attribute__((ext_vector_type(8)))  unsigned int v8u;

#define P_DIM 256
#define U_DIM 512
#define C_DIM 3
#define TN 64
#define TU 64
#define KP2 (P_DIM / 2)   // 128 packed bf16 pairs along P
#define LROW 132          // padded LDS row stride (dwords): stride%64=4 -> conflict-free
#define NTHREADS 256

static constexpr float INV_SQRT_2PI = 0.3989422804014327f;
static constexpr float INV_SQRT_2   = 0.7071067811865476f;
static constexpr float TWO_PI_REF   = 2.0f * 3.14159265359f;   // matches reference PI

__device__ __forceinline__ unsigned int bf16_rne(float f) {
    unsigned int u = __float_as_uint(f);
    return (u + 0x7FFFu + ((u >> 16) & 1u)) >> 16;
}
__device__ __forceinline__ unsigned int pack2(float a, float b) {
    return bf16_rne(a) | (bf16_rne(b) << 16);
}

// Branchless expected-ReLU of N(mu, var); exact ReLU selected when var == 0.
// v_rsq/v_rcp/v_exp based; erf via Abramowitz-Stegun 7.1.26 (|err|<1.5e-7),
// sharing the single exp(-w^2/2) between the pdf and the erf polynomial.
__device__ __forceinline__ float erelu(float mu, float var) {
    float rs = __builtin_amdgcn_rsqf(var);        // inf at var==0 (result discarded)
    float s  = var * rs;                          // sqrt(var)
    float w  = mu * rs;
    float z  = fabsf(w) * INV_SQRT_2;
    float E  = __expf(-z * z);                    // exp(-w^2/2)
    float t  = __builtin_amdgcn_rcpf(fmaf(0.3275911f, z, 1.0f));
    float poly = t * fmaf(t, fmaf(t, fmaf(t, fmaf(t, 1.061405429f, -1.453152027f),
                         1.421413741f), -0.284496736f), 0.254829592f);
    float erfa = 1.0f - poly * E;
    erfa = (w < 0.0f) ? -erfa : erfa;
    float g  = s * E * INV_SQRT_2PI + 0.5f * mu * (1.0f + erfa);  // s*0.5*w*(..) == 0.5*mu*(..)
    float r0 = 0.5f * (mu + fabsf(mu));
    return (var != 0.0f) ? g : r0;
}

__device__ __forceinline__ v16bf mk16(unsigned int a0, unsigned int a1, unsigned int a2,
                                      unsigned int a3, unsigned int a4, unsigned int a5,
                                      unsigned int a6, unsigned int a7) {
    v8u u = {a0, a1, a2, a3, a4, a5, a6, a7};
    return __builtin_bit_cast(v16bf, u);
}

__global__ __launch_bounds__(NTHREADS, 2)
void dense_missing_kernel(const float* __restrict__ x,
                          const float* __restrict__ cmean,
                          const float* __restrict__ cvar,
                          const float* __restrict__ clog,
                          const float* __restrict__ kern,
                          const float* __restrict__ bias,
                          float* __restrict__ out) {
    extern __shared__ unsigned int smem[];
    unsigned int* Xdw  = smem;                    // [TN][LROW]  bf16-pair x tile (NaN -> 0)
    unsigned int* Xmsk = Xdw  + TN * LROW;        // [TN][LROW]  per-half NaN mask (0xFFFF)
    unsigned int* KpM  = Xmsk + TN * LROW;        // [TU][LROW]  bf16-pair kernel (B layout)
    unsigned int* KpV  = KpM  + TU * LROW;        // [TU][LROW]  bf16-pair kernel^2
    unsigned int* muPk = KpV  + TU * LROW;        // [C][KP2]    packed mu
    unsigned int* vaPk = muPk + C_DIM * KP2;      // [C][KP2]    packed var
    float* muT = (float*)(vaPk + C_DIM * KP2);    // [P][C] f32 mu      (loglik)
    float* i2v = muT + P_DIM * C_DIM;             // [P][C] 1/(2 var)
    float* ltb = i2v + P_DIM * C_DIM;             // [P][C] 0.5*log(2 pi var)
    float* llk = ltb + P_DIM * C_DIM;             // [TN][C] loglik accumulators
    float* prb = llk + TN * C_DIM;                // [TN][C] softmax probs
    float* bss = prb + TN * C_DIM;                // [TU]    bias slice

    const int tid = threadIdx.x;
    const int ut  = blockIdx.x & (U_DIM / TU - 1);
    const int nt  = blockIdx.x >> 3;              // U_DIM/TU == 8
    const int n0  = nt * TN;
    const int u0g = ut * TU;

    // ---- Phase A: stage weights / tables -------------------------------
    for (int idx = tid; idx < P_DIM * C_DIM; idx += NTHREADS) {
        float mv = cmean[idx];
        float vv = cvar[idx];
        muT[idx] = mv;
        i2v[idx] = 0.5f / vv;
        ltb[idx] = 0.5f * logf(TWO_PI_REF * vv);
    }
    for (int idx = tid; idx < C_DIM * KP2; idx += NTHREADS) {
        int c = idx / KP2, k2 = idx - c * KP2;
        muPk[idx] = pack2(cmean[(2 * k2) * C_DIM + c], cmean[(2 * k2 + 1) * C_DIM + c]);
        vaPk[idx] = pack2(cvar[(2 * k2) * C_DIM + c],  cvar[(2 * k2 + 1) * C_DIM + c]);
    }
    for (int idx = tid; idx < TU * KP2; idx += NTHREADS) {
        int k2 = idx >> 6;            // idx / TU
        int u  = idx & (TU - 1);
        float g0 = kern[(size_t)(2 * k2) * U_DIM + u0g + u];
        float g1 = kern[(size_t)(2 * k2 + 1) * U_DIM + u0g + u];
        KpM[u * LROW + k2] = pack2(g0, g1);
        KpV[u * LROW + k2] = pack2(g0 * g0, g1 * g1);
    }
    if (tid < TU) bss[tid] = bias[u0g + tid];
    if (tid < TN * C_DIM) llk[tid] = 0.0f;
    __syncthreads();

    // ---- Phase B: x tile -> bf16 + NaN-mask in LDS, loglik partials ----
    {
        int row = tid >> 2;
        int p0  = (tid & 3) * 4;
        const float4* xr4 = reinterpret_cast<const float4*>(x + (size_t)(n0 + row) * P_DIM);
        float q0 = 0.f, q1 = 0.f, q2 = 0.f;
        #pragma unroll
        for (int i = 0; i < 16; ++i) {
            int p = p0 + i * 16;
            float4 v = xr4[p >> 2];
            float e[4] = {v.x, v.y, v.z, v.w};
            unsigned int pk[4], mh[4];
            #pragma unroll
            for (int j = 0; j < 4; ++j) {
                float xv = e[j];
                bool isn = (xv != xv);
                pk[j] = isn ? 0u : bf16_rne(xv);
                mh[j] = isn ? 0xFFFFu : 0u;
                if (!isn) {
                    int pi = (p + j) * C_DIM;
                    float d0 = xv - muT[pi + 0];
                    float d1 = xv - muT[pi + 1];
                    float d2 = xv - muT[pi + 2];
                    q0 -= d0 * d0 * i2v[pi + 0] + ltb[pi + 0];
                    q1 -= d1 * d1 * i2v[pi + 1] + ltb[pi + 1];
                    q2 -= d2 * d2 * i2v[pi + 2] + ltb[pi + 2];
                }
            }
            Xdw[row * LROW + (p >> 1)]      = pk[0] | (pk[1] << 16);
            Xdw[row * LROW + (p >> 1) + 1]  = pk[2] | (pk[3] << 16);
            Xmsk[row * LROW + (p >> 1)]     = mh[0] | (mh[1] << 16);
            Xmsk[row * LROW + (p >> 1) + 1] = mh[2] | (mh[3] << 16);
        }
        atomicAdd(&llk[row * C_DIM + 0], q0);
        atomicAdd(&llk[row * C_DIM + 1], q1);
        atomicAdd(&llk[row * C_DIM + 2], q2);
    }
    __syncthreads();

    // ---- softmax over components per row -------------------------------
    if (tid < TN) {
        float a0 = llk[tid * 3 + 0] + clog[0];
        float a1 = llk[tid * 3 + 1] + clog[1];
        float a2 = llk[tid * 3 + 2] + clog[2];
        float mx = fmaxf(a0, fmaxf(a1, a2));
        float e0 = expf(a0 - mx), e1 = expf(a1 - mx), e2 = expf(a2 - mx);
        float inv = 1.0f / (e0 + e1 + e2);
        prb[tid * 3 + 0] = e0 * inv;
        prb[tid * 3 + 1] = e1 * inv;
        prb[tid * 3 + 2] = e2 * inv;
    }
    __syncthreads();

    // ---- Phase C: WMMA GEMMs, ks-outer / c-inner (B & x frags reused) --
    const int lane = tid & 31;
    const int w    = tid >> 5;
    const int wn   = w & 3, wu = w >> 2;  // 4x2 wave grid
    const int row0 = wn * 16;
    const int u0l  = wu * 32;
    const int ml   = lane & 15, h = lane >> 4;

    const unsigned int* xr  = Xdw  + (row0 + ml) * LROW;
    const unsigned int* mr  = Xmsk + (row0 + ml) * LROW;
    const unsigned int* kr0 = KpM + (u0l + ml) * LROW;
    const unsigned int* kr1 = kr0 + 16 * LROW;
    const unsigned int* vr0 = KpV + (u0l + ml) * LROW;
    const unsigned int* vr1 = vr0 + 16 * LROW;

    const v8f zero = {0.f, 0.f, 0.f, 0.f, 0.f, 0.f, 0.f, 0.f};
    v8f mAcc[C_DIM][2], vAcc[C_DIM][2];
    #pragma unroll
    for (int c = 0; c < C_DIM; ++c) {
        mAcc[c][0] = zero; mAcc[c][1] = zero;
        vAcc[c][0] = zero; vAcc[c][1] = zero;
    }

    #pragma unroll 1
    for (int ks = 0; ks < 8; ++ks) {
        // c-independent fragments: x, NaN-mask (A side), K and K^2 (B side)
        const int pb = ks * 16 + 4 * h;
        uint4 xlo = *reinterpret_cast<const uint4*>(xr + pb);
        uint4 xhi = *reinterpret_cast<const uint4*>(xr + pb + 8);
        uint4 klo = *reinterpret_cast<const uint4*>(mr + pb);
        uint4 khi = *reinterpret_cast<const uint4*>(mr + pb + 8);

        const int kb = ks * 16 + 4 * h;
        uint4 b0l = *reinterpret_cast<const uint4*>(kr0 + kb);
        uint4 b0h = *reinterpret_cast<const uint4*>(kr0 + kb + 8);
        uint4 b1l = *reinterpret_cast<const uint4*>(kr1 + kb);
        uint4 b1h = *reinterpret_cast<const uint4*>(kr1 + kb + 8);
        uint4 q0l = *reinterpret_cast<const uint4*>(vr0 + kb);
        uint4 q0h = *reinterpret_cast<const uint4*>(vr0 + kb + 8);
        uint4 q1l = *reinterpret_cast<const uint4*>(vr1 + kb);
        uint4 q1h = *reinterpret_cast<const uint4*>(vr1 + kb + 8);

        v16bf B0 = mk16(b0l.x, b0l.y, b0l.z, b0l.w, b0h.x, b0h.y, b0h.z, b0h.w);
        v16bf B1 = mk16(b1l.x, b1l.y, b1l.z, b1l.w, b1h.x, b1h.y, b1h.z, b1h.w);
        v16bf Q0 = mk16(q0l.x, q0l.y, q0l.z, q0l.w, q0h.x, q0h.y, q0h.z, q0h.w);
        v16bf Q1 = mk16(q1l.x, q1l.y, q1l.z, q1l.w, q1h.x, q1h.y, q1h.z, q1h.w);

        #pragma unroll
        for (int c = 0; c < C_DIM; ++c) {
            const unsigned int* mup = muPk + c * KP2;
            const unsigned int* vap = vaPk + c * KP2;
            // broadcast pair loads (lane-halves share addresses)
            uint4 ulo = *reinterpret_cast<const uint4*>(mup + pb);
            uint4 uhi = *reinterpret_cast<const uint4*>(mup + pb + 8);
            uint4 wlo = *reinterpret_cast<const uint4*>(vap + pb);
            uint4 whi = *reinterpret_cast<const uint4*>(vap + pb + 8);

            // mean A: (x & ~m) | (mu & m) -> v_bitop3; var A: var & m -> v_and
            v16bf Am = mk16((xlo.x & ~klo.x) | (ulo.x & klo.x),
                            (xlo.y & ~klo.y) | (ulo.y & klo.y),
                            (xlo.z & ~klo.z) | (ulo.z & klo.z),
                            (xlo.w & ~klo.w) | (ulo.w & klo.w),
                            (xhi.x & ~khi.x) | (uhi.x & khi.x),
                            (xhi.y & ~khi.y) | (uhi.y & khi.y),
                            (xhi.z & ~khi.z) | (uhi.z & khi.z),
                            (xhi.w & ~khi.w) | (uhi.w & khi.w));
            v16bf Av = mk16(wlo.x & klo.x, wlo.y & klo.y, wlo.z & klo.z, wlo.w & klo.w,
                            whi.x & khi.x, whi.y & khi.y, whi.z & khi.z, whi.w & khi.w);

            mAcc[c][0] = __builtin_amdgcn_wmma_f32_16x16x32_bf16(false, Am, false, B0, (short)0, mAcc[c][0], false, false);
            mAcc[c][1] = __builtin_amdgcn_wmma_f32_16x16x32_bf16(false, Am, false, B1, (short)0, mAcc[c][1], false, false);
            vAcc[c][0] = __builtin_amdgcn_wmma_f32_16x16x32_bf16(false, Av, false, Q0, (short)0, vAcc[c][0], false, false);
            vAcc[c][1] = __builtin_amdgcn_wmma_f32_16x16x32_bf16(false, Av, false, Q1, (short)0, vAcc[c][1], false, false);
        }
    }

    // ---- fused branchless epilogue: expected_relu + weighted combine ---
    v8f out0 = zero, out1 = zero;
    const float bcol0 = bss[u0l + ml];
    const float bcol1 = bss[u0l + 16 + ml];
    #pragma unroll
    for (int c = 0; c < C_DIM; ++c) {
        #pragma unroll
        for (int e = 0; e < 8; ++e) {
            int rl = row0 + e + 8 * h;           // C/D layout: M = r + 8*(lane>=16)
            float p = prb[rl * C_DIM + c];
            out0[e] += p * erelu(mAcc[c][0][e] + bcol0, vAcc[c][0][e]);
            out1[e] += p * erelu(mAcc[c][1][e] + bcol1, vAcc[c][1][e]);
        }
    }

    #pragma unroll
    for (int e = 0; e < 8; ++e) {
        int rg = n0 + row0 + e + 8 * h;
        out[(size_t)rg * U_DIM + u0g + u0l + ml]      = out0[e];
        out[(size_t)rg * U_DIM + u0g + u0l + 16 + ml] = out1[e];
    }
}

extern "C" void kernel_launch(void* const* d_in, const int* in_sizes, int n_in,
                              void* d_out, int out_size, void* d_ws, size_t ws_size,
                              hipStream_t stream) {
    const float* x  = (const float*)d_in[0];
    const float* cm = (const float*)d_in[1];
    const float* cv = (const float*)d_in[2];
    const float* cl = (const float*)d_in[3];
    const float* kn = (const float*)d_in[4];
    const float* bs = (const float*)d_in[5];
    float* out = (float*)d_out;

    const int N = in_sizes[0] / P_DIM;                 // 65536
    const int grid = (N / TN) * (U_DIM / TU);          // 1024 * 8 = 8192

    const size_t smem_dwords =
        4 * (size_t)TN * LROW                       // Xdw, Xmsk, KpM, KpV (TN==TU)
        + 2 * C_DIM * KP2                           // muPk, vaPk
        + 3 * P_DIM * C_DIM + 2 * TN * C_DIM + TU;  // f32 tables
    const size_t smem_bytes = smem_dwords * 4;      // ~146 KB

    dense_missing_kernel<<<grid, NTHREADS, smem_bytes, stream>>>(x, cm, cv, cl, kn, bs, out);
}
